// AttentionInteractionNetwork_40604620816455
// MI455X (gfx1250) — matrix operand order repaired and checked by reading, not compiled, not run on hardware
//
#include <hip/hip_runtime.h>
#include <hip/hip_bf16.h>
#include <math.h>

// ---------------------------------------------------------------------------
// CDNA5 (gfx1250) fused attention interaction network.
// Big GEMMs run on V_WMMA_F32_16X16X32_BF16 (bf16 in, f32 accum).
// Block-cooperative tile: 4 waves share a 64-row LDS activation tile and
// split N, so each block reads the weights once per 64 rows (quarter the
// L2 weight traffic of the 16-row/wave version while keeping ~100 VGPRs).
// ---------------------------------------------------------------------------

typedef __attribute__((ext_vector_type(16))) __bf16 v16bf;
typedef __attribute__((ext_vector_type(8)))  __bf16 v8bf;
typedef __attribute__((ext_vector_type(8)))  float  v8f;

// ---------------- order-preserving float<->uint encoding for atomicMax ------
__device__ __forceinline__ unsigned encf(float f) {
    unsigned u = __float_as_uint(f);
    return (u & 0x80000000u) ? ~u : (u | 0x80000000u);
}
__device__ __forceinline__ float decf(unsigned u) {
    return __uint_as_float((u & 0x80000000u) ? (u & 0x7FFFFFFFu) : ~u);
}

// ---------------- WMMA K-loop: FOUR 16-row tiles sharing one B fragment -----
// A fragment (ISA 16-bit A 16x32): lane l (m=l&15, g=l>>4) holds two 8-half
// contiguous runs at K = 32*kt + 8g and K = 32*kt + 16 + 8g.
// B fragment (16-bit B 32x16): lane l (n=l&15, g=l>>4) holds two 8-half
// runs at K = 32*kt + 8g and K = 32*kt + 16 + 8g (weights stored transposed [N][K]).
__device__ __forceinline__ void wmma_k4(const __bf16* ldsM, int astr,
                                        int lm, int g, const __bf16* w,
                                        int kTiles, float bias, v8f* acc) {
    union U { v16bf v; v8bf h[2]; };
#pragma unroll
    for (int t = 0; t < 4; ++t)
        acc[t] = (v8f){bias, bias, bias, bias, bias, bias, bias, bias};
    for (int kt = 0; kt < kTiles; ++kt) {
        const int kb = kt * 32;
        U B;
        B.h[0] = *(const v8bf*)(w + kb - g * 8);
        B.h[1] = *(const v8bf*)(w + kb - g * 8 + 16);
#pragma unroll
        for (int t = 0; t < 4; ++t) {
            U A;
            const __bf16* ar = ldsM + (t * 16 + lm) * astr + g * 8 + kb;
            A.h[0] = *(const v8bf*)ar;
            A.h[1] = *(const v8bf*)(ar + 16);
            acc[t] = __builtin_amdgcn_wmma_f32_16x16x32_bf16(
                false, A.v, false, B.v, (short)0, acc[t], false, false);
        }
    }
}

// ---------------- 3-layer MLP core for a 64-row tile (4 waves / block) ------
// ldsA: in [64 x 768] bf16; reused for h2 [64 x 512] bf16; its dead tail
//       (past element 64*512) hosts the LN scratch (psum/psq/stats).
// ldsB: h1 [64 x 512] bf16; reused as f32 [64 x 256] pre-LN output.
// Wave w handles n-tiles [w*8, w*8+8) for GEMM1/2 and [w*4, w*4+4) for GEMM3.
__device__ __forceinline__ void mlp_core64(int tid, __bf16* ldsA, __bf16* ldsB,
                                           const __bf16* wt0, const float* b0,
                                           const __bf16* wt1, const float* b1,
                                           const __bf16* wt2, const float* b2) {
    const int w = tid >> 5, lane = tid & 31;
    const int g = lane >> 4, lm = lane & 15;
    float* psum  = (float*)(ldsA + 64 * 512);   // [64][4] per-wave row sums
    float* psq   = psum + 256;                  // [64][4] per-wave row sumsq
    float* statb = psq + 256;                   // [64][2] {mu, rstd}
    v8f acc[4];

    // GEMM1: [64x768] @ [768x512] + b0 -> SiLU -> ldsB (bf16, stride 512)
    for (int i = 0; i < 8; ++i) {
        const int n = (w * 8 + i) * 16 + lm;
        wmma_k4(ldsA, 768, lm, g, wt0 + (size_t)n * 768 + g * 16, 24, b0[n], acc);
#pragma unroll
        for (int t = 0; t < 4; ++t)
#pragma unroll
            for (int v = 0; v < 8; ++v) {
                float x = acc[t][v];
                ldsB[(t * 16 + v + 8 * g) * 512 + n] =
                    (__bf16)(x / (1.0f + __expf(-x)));
            }
    }
    __syncthreads();

    // GEMM2: [64x512] @ [512x512] + b1 -> SiLU -> ldsA (bf16, stride 512)
    for (int i = 0; i < 8; ++i) {
        const int n = (w * 8 + i) * 16 + lm;
        wmma_k4(ldsB, 512, lm, g, wt1 + (size_t)n * 512 + g * 16, 16, b1[n], acc);
#pragma unroll
        for (int t = 0; t < 4; ++t)
#pragma unroll
            for (int v = 0; v < 8; ++v) {
                float x = acc[t][v];
                ldsA[(t * 16 + v + 8 * g) * 512 + n] =
                    (__bf16)(x / (1.0f + __expf(-x)));
            }
    }
    __syncthreads();

    // GEMM3: [64x512] @ [512x256] + b2 -> f32 buffer + per-wave LN partials
    float* fbuf = (float*)ldsB;
    float rsum[4][8], rss[4][8];
#pragma unroll
    for (int t = 0; t < 4; ++t)
#pragma unroll
        for (int v = 0; v < 8; ++v) { rsum[t][v] = 0.0f; rss[t][v] = 0.0f; }
    for (int i = 0; i < 4; ++i) {
        const int n = (w * 4 + i) * 16 + lm;
        wmma_k4(ldsA, 512, lm, g, wt2 + (size_t)n * 512 + g * 16, 16, b2[n], acc);
#pragma unroll
        for (int t = 0; t < 4; ++t)
#pragma unroll
            for (int v = 0; v < 8; ++v) {
                float x = acc[t][v];
                rsum[t][v] += x; rss[t][v] += x * x;
                fbuf[(t * 16 + v + 8 * g) * 256 + n] = x;
            }
    }
    // intra-wave reduce over the 16 lanes holding the same row, store partials
#pragma unroll
    for (int t = 0; t < 4; ++t)
#pragma unroll
        for (int v = 0; v < 8; ++v) {
            float s = rsum[t][v], q = rss[t][v];
            for (int off = 1; off < 16; off <<= 1) {
                s += __shfl_xor(s, off, 16);
                q += __shfl_xor(q, off, 16);
            }
            if (lm == 0) {
                const int m = t * 16 + v + 8 * g;
                psum[m * 4 + w] = s;
                psq [m * 4 + w] = q;
            }
        }
    __syncthreads();
    // finalize stats: one thread per row
    if (tid < 64) {
        const float s = psum[tid * 4] + psum[tid * 4 + 1]
                      + psum[tid * 4 + 2] + psum[tid * 4 + 3];
        const float q = psq[tid * 4] + psq[tid * 4 + 1]
                      + psq[tid * 4 + 2] + psq[tid * 4 + 3];
        const float mu  = s * (1.0f / 256.0f);
        const float var = q * (1.0f / 256.0f) - mu * mu;
        statb[tid * 2]     = mu;
        statb[tid * 2 + 1] = rsqrtf(var + 1e-5f);
    }
    __syncthreads();
}

// ---------------- weight convert: f32 [K][N] -> bf16 transposed [N][K] ------
__global__ void transpose_to_bf16(const float* __restrict__ W,
                                  __bf16* __restrict__ Wt, int K, int N) {
    int idx = blockIdx.x * blockDim.x + threadIdx.x;
    if (idx >= K * N) return;
    int n = idx / K, k = idx - n * K;
    Wt[(size_t)n * K + k] = (__bf16)W[(size_t)k * N + n];
}

// ---------------- attention logits: one wave per edge -----------------------
__global__ __launch_bounds__(256) void attn_logits_kernel(
    const float* __restrict__ edges,
    const float* __restrict__ rw, const float* __restrict__ rb,
    const float* __restrict__ sw, const float* __restrict__ sb,
    float* __restrict__ logr, float* __restrict__ logs, int nE) {
    const int e = (blockIdx.x * 256 + threadIdx.x) >> 5;
    const int lane = threadIdx.x & 31;
    if (e >= nE) return;
    const float* ep = edges + (size_t)e * 256 + lane * 8;
    const float4 a0 = *(const float4*)ep;
    const float4 a1 = *(const float4*)(ep + 4);
    const float4 r0 = *(const float4*)(rw + lane * 8);
    const float4 r1 = *(const float4*)(rw + lane * 8 + 4);
    const float4 s0 = *(const float4*)(sw + lane * 8);
    const float4 s1 = *(const float4*)(sw + lane * 8 + 4);
    float dr = a0.x * r0.x + a0.y * r0.y + a0.z * r0.z + a0.w * r0.w
             + a1.x * r1.x + a1.y * r1.y + a1.z * r1.z + a1.w * r1.w;
    float ds = a0.x * s0.x + a0.y * s0.y + a0.z * s0.z + a0.w * s0.w
             + a1.x * s1.x + a1.y * s1.y + a1.z * s1.z + a1.w * s1.w;
    for (int off = 16; off > 0; off >>= 1) {
        dr += __shfl_xor(dr, off, 32);
        ds += __shfl_xor(ds, off, 32);
    }
    if (lane == 0) { logr[e] = dr + rb[0]; logs[e] = ds + sb[0]; }
}

__global__ void seg_max_kernel(const float* __restrict__ logr,
                               const float* __restrict__ logs,
                               const int* __restrict__ senders,
                               const int* __restrict__ receivers,
                               unsigned* __restrict__ maxr,
                               unsigned* __restrict__ maxs, int nE) {
    int t = blockIdx.x * blockDim.x + threadIdx.x;
    if (t >= nE) return;
    atomicMax(maxr + receivers[t], encf(logr[t]));
    atomicMax(maxs + senders[t], encf(logs[t]));
}

__global__ void seg_expsum_kernel(float* __restrict__ logr, float* __restrict__ logs,
                                  const int* __restrict__ senders,
                                  const int* __restrict__ receivers,
                                  const unsigned* __restrict__ maxr,
                                  const unsigned* __restrict__ maxs,
                                  float* __restrict__ sumr, float* __restrict__ sums,
                                  int nE) {
    int t = blockIdx.x * blockDim.x + threadIdx.x;
    if (t >= nE) return;
    const int r = receivers[t], s = senders[t];
    const float er = __expf(logr[t] - decf(maxr[r]));
    const float es = __expf(logs[t] - decf(maxs[s]));
    logr[t] = er; logs[t] = es;
    atomicAdd(sumr + r, er);
    atomicAdd(sums + s, es);
}

__global__ void attn_norm_kernel(float* __restrict__ logr, float* __restrict__ logs,
                                 const int* __restrict__ senders,
                                 const int* __restrict__ receivers,
                                 const float* __restrict__ sumr,
                                 const float* __restrict__ sums, int nE) {
    int t = blockIdx.x * blockDim.x + threadIdx.x;
    if (t >= nE) return;
    logr[t] = logr[t] / sumr[receivers[t]];
    logs[t] = logs[t] / sums[senders[t]];
}

// ---------------- fused edge MLP + LN + residual + attention scatter --------
__global__ __launch_bounds__(128) void edge_mlp_kernel(
    const float* __restrict__ nodes, const float* __restrict__ edges,
    const int* __restrict__ senders, const int* __restrict__ receivers,
    const float* __restrict__ attn_s, const float* __restrict__ attn_r,
    const __bf16* __restrict__ wt0, const float* __restrict__ b0,
    const __bf16* __restrict__ wt1, const float* __restrict__ b1,
    const __bf16* __restrict__ wt2, const float* __restrict__ b2,
    const float* __restrict__ ln_g, const float* __restrict__ ln_b,
    float* __restrict__ out_edges, float* __restrict__ sent_agg,
    float* __restrict__ recv_agg, int nE) {
    __shared__ __align__(16) __bf16 ldsA[64 * 768];   // 96 KB
    __shared__ __align__(16) __bf16 ldsB[64 * 512];   // 64 KB
    const int tid = threadIdx.x;
    const int e0 = blockIdx.x * 64;

    // stage edge_in = [edges[e] | nodes[snd] | nodes[rcv]] as bf16
    for (int idx = tid; idx < 64 * 192; idx += 128) {  // 192 float4 / row
        const int row = idx / 192;
        const int col = (idx % 192) * 4;
        int e = e0 + row; if (e >= nE) e = nE - 1;
        const float* src;
        if (col < 256)      src = edges + (size_t)e * 256 + col;
        else if (col < 512) src = nodes + (size_t)senders[e] * 256 + (col - 256);
        else                src = nodes + (size_t)receivers[e] * 256 + (col - 512);
        const float4 val = *(const float4*)src;
        __bf16* dst = ldsA + row * 768 + col;
        dst[0] = (__bf16)val.x; dst[1] = (__bf16)val.y;
        dst[2] = (__bf16)val.z; dst[3] = (__bf16)val.w;
    }
    __syncthreads();

    mlp_core64(tid, ldsA, ldsB, wt0, b0, wt1, b1, wt2, b2);

    // LN + residual write + attention-weighted scatter-add
    const float* fbuf  = (const float*)ldsB;
    const float* statb = (const float*)(ldsA + 64 * 512) + 512;
    for (int idx = tid; idx < 64 * 64; idx += 128) {   // 64 float4 / row
        const int m = idx >> 6;
        const int c = (idx & 63) * 4;
        const int e = e0 + m;
        if (e >= nE) continue;
        const float mu = statb[m * 2], rstd = statb[m * 2 + 1];
        const float4 h  = *(const float4*)(fbuf + m * 256 + c);
        const float4 gg = *(const float4*)(ln_g + c);
        const float4 bb = *(const float4*)(ln_b + c);
        float4 u;
        u.x = (h.x - mu) * rstd * gg.x + bb.x;
        u.y = (h.y - mu) * rstd * gg.y + bb.y;
        u.z = (h.z - mu) * rstd * gg.z + bb.z;
        u.w = (h.w - mu) * rstd * gg.w + bb.w;
        const float4 ein = *(const float4*)(edges + (size_t)e * 256 + c);
        float4 o = {ein.x + u.x, ein.y + u.y, ein.z + u.z, ein.w + u.w};
        *(float4*)(out_edges + (size_t)e * 256 + c) = o;
        const int s = senders[e], r = receivers[e];
        const float as = attn_s[e], ar = attn_r[e];
        float* sa = sent_agg + (size_t)s * 256 + c;
        float* ra = recv_agg + (size_t)r * 256 + c;
        atomicAdd(sa + 0, u.x * as); atomicAdd(sa + 1, u.y * as);
        atomicAdd(sa + 2, u.z * as); atomicAdd(sa + 3, u.w * as);
        atomicAdd(ra + 0, u.x * ar); atomicAdd(ra + 1, u.y * ar);
        atomicAdd(ra + 2, u.z * ar); atomicAdd(ra + 3, u.w * ar);
    }
}

// ---------------- fused node MLP + LN + residual ----------------------------
__global__ __launch_bounds__(128) void node_mlp_kernel(
    const float* __restrict__ nodes, const float* __restrict__ recv_agg,
    const float* __restrict__ sent_agg,
    const __bf16* __restrict__ wt0, const float* __restrict__ b0,
    const __bf16* __restrict__ wt1, const float* __restrict__ b1,
    const __bf16* __restrict__ wt2, const float* __restrict__ b2,
    const float* __restrict__ ln_g, const float* __restrict__ ln_b,
    float* __restrict__ out_nodes, int nN) {
    __shared__ __align__(16) __bf16 ldsA[64 * 768];
    __shared__ __align__(16) __bf16 ldsB[64 * 512];
    const int tid = threadIdx.x;
    const int v0 = blockIdx.x * 64;

    // stage node_in = [nodes[v] | recv_agg[v] | sent_agg[v]] as bf16
    for (int idx = tid; idx < 64 * 192; idx += 128) {
        const int row = idx / 192;
        const int col = (idx % 192) * 4;
        int nd = v0 + row; if (nd >= nN) nd = nN - 1;
        const float* src;
        if (col < 256)      src = nodes    + (size_t)nd * 256 + col;
        else if (col < 512) src = recv_agg + (size_t)nd * 256 + (col - 256);
        else                src = sent_agg + (size_t)nd * 256 + (col - 512);
        const float4 val = *(const float4*)src;
        __bf16* dst = ldsA + row * 768 + col;
        dst[0] = (__bf16)val.x; dst[1] = (__bf16)val.y;
        dst[2] = (__bf16)val.z; dst[3] = (__bf16)val.w;
    }
    __syncthreads();

    mlp_core64(tid, ldsA, ldsB, wt0, b0, wt1, b1, wt2, b2);

    const float* fbuf  = (const float*)ldsB;
    const float* statb = (const float*)(ldsA + 64 * 512) + 512;
    for (int idx = tid; idx < 64 * 64; idx += 128) {
        const int m = idx >> 6;
        const int c = (idx & 63) * 4;
        const int nd = v0 + m;
        if (nd >= nN) continue;
        const float mu = statb[m * 2], rstd = statb[m * 2 + 1];
        const float4 h   = *(const float4*)(fbuf + m * 256 + c);
        const float4 gg  = *(const float4*)(ln_g + c);
        const float4 bb  = *(const float4*)(ln_b + c);
        const float4 xin = *(const float4*)(nodes + (size_t)nd * 256 + c);
        float4 o;
        o.x = xin.x + (h.x - mu) * rstd * gg.x + bb.x;
        o.y = xin.y + (h.y - mu) * rstd * gg.y + bb.y;
        o.z = xin.z + (h.z - mu) * rstd * gg.z + bb.z;
        o.w = xin.w + (h.w - mu) * rstd * gg.w + bb.w;
        *(float4*)(out_nodes + (size_t)nd * 256 + c) = o;
    }
}

// ---------------------------------------------------------------------------
extern "C" void kernel_launch(void* const* d_in, const int* in_sizes, int n_in,
                              void* d_out, int out_size, void* d_ws, size_t ws_size,
                              hipStream_t stream) {
    const float* nodes     = (const float*)d_in[0];
    const float* edges     = (const float*)d_in[1];
    const int*   senders   = (const int*)d_in[2];
    const int*   receivers = (const int*)d_in[3];
    const float* eW0 = (const float*)d_in[4];  const float* eb0 = (const float*)d_in[5];
    const float* eW1 = (const float*)d_in[6];  const float* eb1 = (const float*)d_in[7];
    const float* eW2 = (const float*)d_in[8];  const float* eb2 = (const float*)d_in[9];
    const float* elg = (const float*)d_in[10]; const float* elb = (const float*)d_in[11];
    const float* nW0 = (const float*)d_in[12]; const float* nb0 = (const float*)d_in[13];
    const float* nW1 = (const float*)d_in[14]; const float* nb1 = (const float*)d_in[15];
    const float* nW2 = (const float*)d_in[16]; const float* nb2 = (const float*)d_in[17];
    const float* nlg = (const float*)d_in[18]; const float* nlb = (const float*)d_in[19];
    const float* rw  = (const float*)d_in[20]; const float* rb  = (const float*)d_in[21];
    const float* sw  = (const float*)d_in[22]; const float* sb  = (const float*)d_in[23];

    const int N = in_sizes[0] / 256;
    const int E = in_sizes[1] / 256;

    // ---- workspace carve (zeroed region first, then scratch, then weights)
    char* base = (char*)d_ws;
    size_t off = 0;
    auto carve = [&](size_t bytes) -> void* {
        void* p = base + off;
        off = (off + bytes + 255) & ~(size_t)255;
        return p;
    };
    unsigned* maxr     = (unsigned*)carve((size_t)N * 4);
    unsigned* maxs     = (unsigned*)carve((size_t)N * 4);
    float*    sumr     = (float*)carve((size_t)N * 4);
    float*    sums     = (float*)carve((size_t)N * 4);
    float*    sent_agg = (float*)carve((size_t)N * 256 * 4);
    float*    recv_agg = (float*)carve((size_t)N * 256 * 4);
    const size_t zbytes = off;                 // everything above must be zero
    float*  logr  = (float*)carve((size_t)E * 4);
    float*  logs  = (float*)carve((size_t)E * 4);
    __bf16* wt_e0 = (__bf16*)carve((size_t)768 * 512 * 2);
    __bf16* wt_e1 = (__bf16*)carve((size_t)512 * 512 * 2);
    __bf16* wt_e2 = (__bf16*)carve((size_t)512 * 256 * 2);
    __bf16* wt_n0 = (__bf16*)carve((size_t)768 * 512 * 2);
    __bf16* wt_n1 = (__bf16*)carve((size_t)512 * 512 * 2);
    __bf16* wt_n2 = (__bf16*)carve((size_t)512 * 256 * 2);

    hipMemsetAsync(d_ws, 0, zbytes, stream);

    auto tr = [&](const float* W, __bf16* Wt, int K, int Ncols) {
        const int elems = K * Ncols;
        transpose_to_bf16<<<(elems + 255) / 256, 256, 0, stream>>>(W, Wt, K, Ncols);
    };
    tr(eW0, wt_e0, 768, 512); tr(eW1, wt_e1, 512, 512); tr(eW2, wt_e2, 512, 256);
    tr(nW0, wt_n0, 768, 512); tr(nW1, wt_n1, 512, 512); tr(nW2, wt_n2, 512, 256);

    attn_logits_kernel<<<(E + 7) / 8, 256, 0, stream>>>(edges, rw, rb, sw, sb,
                                                        logr, logs, E);
    seg_max_kernel<<<(E + 255) / 256, 256, 0, stream>>>(logr, logs, senders,
                                                        receivers, maxr, maxs, E);
    seg_expsum_kernel<<<(E + 255) / 256, 256, 0, stream>>>(logr, logs, senders,
                                                           receivers, maxr, maxs,
                                                           sumr, sums, E);
    attn_norm_kernel<<<(E + 255) / 256, 256, 0, stream>>>(logr, logs, senders,
                                                          receivers, sumr, sums, E);

    float* out_nodes = (float*)d_out;
    float* out_edges = out_nodes + (size_t)N * 256;

    edge_mlp_kernel<<<(E + 63) / 64, 128, 0, stream>>>(
        nodes, edges, senders, receivers, /*attn_s=*/logs, /*attn_r=*/logr,
        wt_e0, eb0, wt_e1, eb1, wt_e2, eb2, elg, elb,
        out_edges, sent_agg, recv_agg, E);

    node_mlp_kernel<<<(N + 63) / 64, 128, 0, stream>>>(
        nodes, recv_agg, sent_agg,
        wt_n0, nb0, wt_n1, nb1, wt_n2, nb2, nlg, nlb,
        out_nodes, N);
}
